// CustomMultiHeadAttention_4973572129094
// MI455X (gfx1250) — matrix operand order repaired and checked by reading, not compiled, not run on hardware
//
#include <hip/hip_runtime.h>
#include <hip/hip_bf16.h>

// ---------------------------------------------------------------------------
// MHA forward for MI455X (gfx1250, wave32, WMMA).
// fp32 emulated on bf16 matrix cores via hi/lo split (bf16x3):
//   x = hi + lo;  x*w ~= hi*hi + hi*lo + lo*hi   (fp32 accumulate)
// Weights, Q/K/V and ctx are pre-split into bf16 hi/lo arrays on store; fp32
// A operands (X, attn) are split into hi/lo in the GEMM inner loop on load.
// ---------------------------------------------------------------------------

#define HIDDEN 768
#define HEADS  12
#define DEPTH  64
#define BATCH  2
#define SEQ    2048
#define ROWS   (BATCH * SEQ)          // 4096
#define NPROJ  (ROWS * HIDDEN)        // 3145728 elems per [B*S, H] buffer
#define NW     (HIDDEN * HIDDEN)      // 589824 elems per weight matrix

typedef __attribute__((ext_vector_type(16))) __bf16 v16bf;
typedef __attribute__((ext_vector_type(8)))  __bf16 v8bf;
typedef __attribute__((ext_vector_type(8)))  float  v8f;

__device__ __forceinline__ __bf16 bf16_hi(float x) { return (__bf16)x; }
__device__ __forceinline__ __bf16 bf16_lo(float x, __bf16 hi) {
    return (__bf16)(x - (float)hi);
}

// acc += A*B with split operands; drop lo*lo (below fp32 accumulator noise).
__device__ __forceinline__ v8f wmma3(const v16bf& ah, const v16bf& al,
                                     const v16bf& bh, const v16bf& bl, v8f acc) {
    acc = __builtin_amdgcn_wmma_f32_16x16x32_bf16(false, al, false, bh,
                                                  (short)0, acc, false, false);
    acc = __builtin_amdgcn_wmma_f32_16x16x32_bf16(false, ah, false, bl,
                                                  (short)0, acc, false, false);
    acc = __builtin_amdgcn_wmma_f32_16x16x32_bf16(false, ah, false, bh,
                                                  (short)0, acc, false, false);
    return acc;
}

// ---------------------------------------------------------------------------
// GEMM core: one wave computes a 16 x (NT*16) strip of C = scale*(A@B + bias).
//   A logical (m,k): fp32 (A_SPLIT=0) or pre-split bf16 hi/lo (A_SPLIT=1)
//   B logical (k,n): B_T=1 -> stored [N][K] (weights / K^T), contiguous in k
//                    B_T=0 -> stored [K][N] (V), strided in k
//   C: C_SPLIT=1 -> store bf16 hi/lo pair;  C_SPLIT=0 -> store fp32
//
// ISA lane layouts (CDNA5 ISA, 16-bit 16x16x32, wave32):
//   A elem j (lane): K = (j<8 ? j : j+8) + 8*(lane>=16), row M = lane&15
//   B elem j (lane): K = (j<8 ? j : j+8) + 8*(lane>=16), col N = lane&15
//   C vgpr r (lane): M = r + 8*(lane>=16),               col N = lane&15
// ---------------------------------------------------------------------------
template <int NT, bool A_SPLIT, bool B_T, bool C_SPLIT>
__device__ __forceinline__ void mma_core(
    const float*  __restrict__ Af32,
    const __bf16* __restrict__ Ahi, const __bf16* __restrict__ Alo,
    long lda, int m0,
    const __bf16* __restrict__ Bhi, const __bf16* __restrict__ Blo,
    long ldb, int n0,
    float* __restrict__ Cf32, __bf16* __restrict__ Chi, __bf16* __restrict__ Clo,
    long ldc, int Kd, float scale, const float* __restrict__ bias) {

    const int lane = threadIdx.x & 31;
    const int half = lane >> 4;      // 0: lanes 0-15, 1: lanes 16-31
    const int mr   = lane & 15;      // M row for A operand, N col for B/C

    v8f acc[NT];
#pragma unroll
    for (int t = 0; t < NT; ++t) {
        const float bi = bias ? bias[n0 + 16 * t + mr] : 0.0f;
#pragma unroll
        for (int r = 0; r < 8; ++r) acc[t][r] = bi;
    }

    const float*  arow_f = A_SPLIT ? nullptr : (Af32 + (long)(m0 + mr) * lda);
    const __bf16* arow_h = A_SPLIT ? (Ahi + (long)(m0 + mr) * lda) : nullptr;
    const __bf16* arow_l = A_SPLIT ? (Alo + (long)(m0 + mr) * lda) : nullptr;

    for (int k0 = 0; k0 < Kd; k0 += 32) {
        // ---- A operand: two contiguous 8-elem K chunks per lane ----
        v16bf ah, al;
        if (A_SPLIT) {
            const v8bf h0 = *(const v8bf*)(arow_h + k0 + 8 * half);
            const v8bf h1 = *(const v8bf*)(arow_h + k0 + 16 + 8 * half);
            const v8bf l0 = *(const v8bf*)(arow_l + k0 + 8 * half);
            const v8bf l1 = *(const v8bf*)(arow_l + k0 + 16 + 8 * half);
#pragma unroll
            for (int j = 0; j < 8; ++j) {
                ah[j] = h0[j]; ah[j + 8] = h1[j];
                al[j] = l0[j]; al[j + 8] = l1[j];
            }
        } else {
            const float* a0 = arow_f + k0 + 8 * half;
#pragma unroll
            for (int j = 0; j < 8; ++j) {
                const float x0 = a0[j], x1 = a0[16 + j];
                const __bf16 h0 = bf16_hi(x0), h1 = bf16_hi(x1);
                ah[j]     = h0; al[j]     = bf16_lo(x0, h0);
                ah[j + 8] = h1; al[j + 8] = bf16_lo(x1, h1);
            }
        }

        // ---- B operand per N-tile, then accumulate ----
#pragma unroll
        for (int t = 0; t < NT; ++t) {
            const int nc = n0 + 16 * t + mr;
            v16bf bh, bl;
            if (B_T) {
                const __bf16* bh0 = Bhi + (long)nc * ldb + k0 + 8 * half;
                const __bf16* bl0 = Blo + (long)nc * ldb + k0 + 8 * half;
                const v8bf h0 = *(const v8bf*)(bh0);
                const v8bf h1 = *(const v8bf*)(bh0 + 16);
                const v8bf l0 = *(const v8bf*)(bl0);
                const v8bf l1 = *(const v8bf*)(bl0 + 16);
#pragma unroll
                for (int j = 0; j < 8; ++j) {
                    bh[j] = h0[j]; bh[j + 8] = h1[j];
                    bl[j] = l0[j]; bl[j + 8] = l1[j];
                }
                __builtin_prefetch(bh0 + 256, 0, 0);   // global_prefetch_b8
            } else {
#pragma unroll
                for (int j = 0; j < 16; ++j) {
                    const long kk = k0 + 8 * half + (j < 8 ? j : j + 8);
                    bh[j] = Bhi[kk * ldb + nc];
                    bl[j] = Blo[kk * ldb + nc];
                }
            }
            acc[t] = wmma3(ah, al, bh, bl, acc[t]);
        }
    }

    // ---- store ----
#pragma unroll
    for (int t = 0; t < NT; ++t) {
        const int nc = n0 + 16 * t + mr;
#pragma unroll
        for (int r = 0; r < 8; ++r) {
            const long idx = (long)(m0 + r + 8 * half) * ldc + nc;
            const float v = acc[t][r] * scale;
            if (C_SPLIT) {
                const __bf16 h = bf16_hi(v);
                Chi[idx] = h;
                Clo[idx] = bf16_lo(v, h);
            } else {
                Cf32[idx] = v;
            }
        }
    }
}

// ---------------------------------------------------------------------------
// Elementwise fp32 -> (bf16 hi, bf16 lo) split (for weight matrices)
// ---------------------------------------------------------------------------
__global__ void __launch_bounds__(256)
split_kernel(const float* __restrict__ X, __bf16* __restrict__ hi,
             __bf16* __restrict__ lo, int n) {
    const int i = blockIdx.x * 256 + threadIdx.x;
    if (i < n) {
        const float x = X[i];
        const __bf16 h = bf16_hi(x);
        hi[i] = h;
        lo[i] = bf16_lo(x, h);
    }
}

// ---------------------------------------------------------------------------
// Projection: [4096,768] = X(fp32) @ W^T(split) + b  -> split bf16 output
// grid (3, 256), block 128: wave strip = 16x64, block covers 256 cols
// ---------------------------------------------------------------------------
__global__ void __launch_bounds__(128)
proj_kernel(const float* __restrict__ X,
            const __bf16* __restrict__ Whi, const __bf16* __restrict__ Wlo,
            const float* __restrict__ bias,
            __bf16* __restrict__ Chi, __bf16* __restrict__ Clo) {
    const int wave = threadIdx.x >> 5;
    const int m0 = blockIdx.y * 16;
    const int n0 = (blockIdx.x * 4 + wave) * 64;
    mma_core<4, false, true, true>(X, nullptr, nullptr, HIDDEN, m0,
                                   Whi, Wlo, HIDDEN, n0,
                                   nullptr, Chi, Clo, HIDDEN,
                                   HIDDEN, 1.0f, bias);
}

// ---------------------------------------------------------------------------
// Output projection: out(fp32) = Ctx(split) @ Wo^T(split) + bo
// grid (3, 256), block 128
// ---------------------------------------------------------------------------
__global__ void __launch_bounds__(128)
oproj_kernel(const __bf16* __restrict__ Ahi, const __bf16* __restrict__ Alo,
             const __bf16* __restrict__ Whi, const __bf16* __restrict__ Wlo,
             const float* __restrict__ bias, float* __restrict__ C) {
    const int wave = threadIdx.x >> 5;
    const int m0 = blockIdx.y * 16;
    const int n0 = (blockIdx.x * 4 + wave) * 64;
    mma_core<4, true, true, false>(nullptr, Ahi, Alo, HIDDEN, m0,
                                   Whi, Wlo, HIDDEN, n0,
                                   C, nullptr, nullptr, HIDDEN,
                                   HIDDEN, 1.0f, bias);
}

// ---------------------------------------------------------------------------
// Scores: attn_raw[b,h,q,k] = (Q_h @ K_h^T)/8, all-bf16 operands, fp32 out
// grid (8, 128, 24), block 128: block covers 256 key columns
// ---------------------------------------------------------------------------
__global__ void __launch_bounds__(128)
scores_kernel(const __bf16* __restrict__ Qhi, const __bf16* __restrict__ Qlo,
              const __bf16* __restrict__ Khi, const __bf16* __restrict__ Klo,
              float* __restrict__ attn) {
    const int wave = threadIdx.x >> 5;
    const int bh = blockIdx.z;
    const int b = bh / HEADS, h = bh % HEADS;
    const long off = (long)b * SEQ * HIDDEN + h * DEPTH;
    float* Cbase = attn + (long)bh * SEQ * SEQ;
    const int m0 = blockIdx.y * 16;
    const int n0 = (blockIdx.x * 4 + wave) * 64;
    mma_core<4, true, true, false>(nullptr, Qhi + off, Qlo + off, HIDDEN, m0,
                                   Khi + off, Klo + off, HIDDEN, n0,
                                   Cbase, nullptr, nullptr, SEQ,
                                   DEPTH, 0.125f, nullptr);
}

// ---------------------------------------------------------------------------
// In-place row softmax over 2048 columns. One block per row.
// ---------------------------------------------------------------------------
__global__ void __launch_bounds__(256)
softmax_kernel(float* __restrict__ attn) {
    float* p = attn + (long)blockIdx.x * SEQ;
    const int t = threadIdx.x;
    __shared__ float red[256];

    float v[8];
    float mx = -INFINITY;
#pragma unroll
    for (int i = 0; i < 8; ++i) {
        v[i] = p[t + 256 * i];
        mx = fmaxf(mx, v[i]);
    }
    red[t] = mx;
    __syncthreads();
    for (int s = 128; s > 0; s >>= 1) {
        if (t < s) red[t] = fmaxf(red[t], red[t + s]);
        __syncthreads();
    }
    mx = red[0];
    __syncthreads();

    float sum = 0.0f;
#pragma unroll
    for (int i = 0; i < 8; ++i) {
        v[i] = __expf(v[i] - mx);
        sum += v[i];
    }
    red[t] = sum;
    __syncthreads();
    for (int s = 128; s > 0; s >>= 1) {
        if (t < s) red[t] += red[t + s];
        __syncthreads();
    }
    const float inv = 1.0f / red[0];
#pragma unroll
    for (int i = 0; i < 8; ++i) p[t + 256 * i] = v[i] * inv;
}

// ---------------------------------------------------------------------------
// Context: ctx[b,s,h*64+d] = attn(fp32) @ V_h(split, strided) -> split out
// grid (1, 32, 24), block 128: wave -> one 16-row m-tile, all 64 head cols
// ---------------------------------------------------------------------------
__global__ void __launch_bounds__(128)
ctx_kernel(const float* __restrict__ attn,
           const __bf16* __restrict__ Vhi, const __bf16* __restrict__ Vlo,
           __bf16* __restrict__ Chi, __bf16* __restrict__ Clo) {
    const int wave = threadIdx.x >> 5;
    const int bh = blockIdx.z;
    const int b = bh / HEADS, h = bh % HEADS;
    const long off = (long)b * SEQ * HIDDEN + h * DEPTH;
    const float* Abase = attn + (long)bh * SEQ * SEQ;
    const int m0 = (blockIdx.y * 4 + wave) * 16;
    mma_core<4, false, false, true>(Abase, nullptr, nullptr, SEQ, m0,
                                    Vhi + off, Vlo + off, HIDDEN, 0,
                                    nullptr, Chi + off, Clo + off, HIDDEN,
                                    SEQ, 1.0f, nullptr);
}

// ---------------------------------------------------------------------------
extern "C" void kernel_launch(void* const* d_in, const int* in_sizes, int n_in,
                              void* d_out, int out_size, void* d_ws, size_t ws_size,
                              hipStream_t stream) {
    (void)in_sizes; (void)n_in; (void)out_size; (void)ws_size;
    const float* q  = (const float*)d_in[0];
    const float* k  = (const float*)d_in[1];
    const float* v  = (const float*)d_in[2];
    const float* Wq = (const float*)d_in[3];
    const float* bq = (const float*)d_in[4];
    const float* Wk = (const float*)d_in[5];
    const float* bk = (const float*)d_in[6];
    const float* Wv = (const float*)d_in[7];
    const float* bv = (const float*)d_in[8];
    const float* Wo = (const float*)d_in[9];
    const float* bo = (const float*)d_in[10];

    float* out  = (float*)d_out;          // [4096, 768]
    float* attn = out + (long)NPROJ;      // [24, 2048, 2048]

    // Workspace layout (bf16 elems): 8 weight halves + 8 activation halves
    __bf16* w = (__bf16*)d_ws;            // needs ~60 MB
    __bf16* WqHi = w;                 __bf16* WqLo = WqHi + NW;
    __bf16* WkHi = WqLo + NW;         __bf16* WkLo = WkHi + NW;
    __bf16* WvHi = WkLo + NW;         __bf16* WvLo = WvHi + NW;
    __bf16* WoHi = WvLo + NW;         __bf16* WoLo = WoHi + NW;
    __bf16* QHi  = WoLo + NW;         __bf16* QLo  = QHi + (long)NPROJ;
    __bf16* KHi  = QLo + (long)NPROJ; __bf16* KLo  = KHi + (long)NPROJ;
    __bf16* VHi  = KLo + (long)NPROJ; __bf16* VLo  = VHi + (long)NPROJ;
    __bf16* CxHi = VLo + (long)NPROJ; __bf16* CxLo = CxHi + (long)NPROJ;

    const dim3 blk(128);
    const dim3 gsplit(NW / 256);
    const dim3 gproj(HIDDEN / 256, ROWS / 16);         // (3, 256)

    split_kernel<<<gsplit, dim3(256), 0, stream>>>(Wq, WqHi, WqLo, NW);
    split_kernel<<<gsplit, dim3(256), 0, stream>>>(Wk, WkHi, WkLo, NW);
    split_kernel<<<gsplit, dim3(256), 0, stream>>>(Wv, WvHi, WvLo, NW);
    split_kernel<<<gsplit, dim3(256), 0, stream>>>(Wo, WoHi, WoLo, NW);

    proj_kernel<<<gproj, blk, 0, stream>>>(q, WqHi, WqLo, bq, QHi, QLo);
    proj_kernel<<<gproj, blk, 0, stream>>>(k, WkHi, WkLo, bk, KHi, KLo);
    proj_kernel<<<gproj, blk, 0, stream>>>(v, WvHi, WvLo, bv, VHi, VLo);

    scores_kernel<<<dim3(SEQ / 256, SEQ / 16, BATCH * HEADS), blk, 0, stream>>>(
        QHi, QLo, KHi, KLo, attn);
    softmax_kernel<<<dim3(BATCH * HEADS * SEQ), dim3(256), 0, stream>>>(attn);
    ctx_kernel<<<dim3(1, SEQ / 64, BATCH * HEADS), blk, 0, stream>>>(
        attn, VHi, VLo, CxHi, CxLo);

    oproj_kernel<<<gproj, blk, 0, stream>>>(CxHi, CxLo, WoHi, WoLo, bo, out);
}
